// MultiHeadAttentionBlock_39754217292477
// MI455X (gfx1250) — compile-verified, matrix operand order checked by reading, not run on hardware
//
#include <hip/hip_runtime.h>

// ---------------------------------------------------------------------------
// MHA block for gfx1250 (MI455X). fp32 in/out, bf16 WMMA internally.
// B=2, S=2048, D=1024, H=16, Dk=64.
//
// Roofline: ~69 GFLOP, ~84 MB unique data (L2-resident). GEMMs are WMMA-
// bound. Attention K/V re-streaming was the bottleneck (~2 GB L2 traffic with
// per-wave streaming); now 8 waves per block share each 32-key K/V tile via
// LDS, staged by the Tensor Data Mover (tensor_load_to_lds, double-buffered,
// s_wait_tensorcnt), cutting K/V traffic 8x to ~256 MB.
// ---------------------------------------------------------------------------

typedef __attribute__((ext_vector_type(16))) __bf16       v16bf;
typedef __attribute__((ext_vector_type(8)))  float        v8f;
typedef __attribute__((ext_vector_type(8)))  unsigned int v8u;
typedef __attribute__((ext_vector_type(4)))  unsigned int u32x4;
typedef __attribute__((ext_vector_type(8)))  int          i32x8;
typedef __attribute__((ext_vector_type(4)))  int          i32x4;

#define SLEN  2048
#define DMODEL 1024
#define NHEAD 16
#define DK    64
#define MROWS 4096   // B*S

__device__ __forceinline__ unsigned short f2bf(float f) {
    unsigned int u = __float_as_uint(f);
    u += 0x7FFFu + ((u >> 16) & 1u);   // round-to-nearest-even
    return (unsigned short)(u >> 16);
}

__device__ __forceinline__ v16bf load16(const unsigned short* p) {
    v8u t = *(const v8u*)p;
    return __builtin_bit_cast(v16bf, t);
}

__device__ __forceinline__ v8f wmma_bf16(v16bf a, v16bf b, v8f c) {
    return __builtin_amdgcn_wmma_f32_16x16x32_bf16(
        false, a, false, b, (short)0, c, false, false);
}

// ---------------------------------------------------------------------------
// fp32 -> bf16 conversion
// ---------------------------------------------------------------------------
__global__ void cvt_bf16_kernel(const float* __restrict__ in,
                                unsigned short* __restrict__ out, int n) {
    int i = blockIdx.x * blockDim.x + threadIdx.x;
    int stride = gridDim.x * blockDim.x;
    for (; i < n; i += stride) out[i] = f2bf(in[i]);
}

// ---------------------------------------------------------------------------
// Projection GEMM: Y = X[4096,1024] * W^T[1024,1024] + bias
// Per wave: 32(M) x 64(N) macro tile, K-steps of 32 -> 8 WMMAs / step.
// mode 0: Q -> [B*H, S, Dk] bf16, scaled by 1/sqrt(Dk)
// mode 1: K -> [B*H, S, Dk] bf16
// mode 2: V -> [B*H, Dk, S] bf16 (transposed for the PV B-operand)
// ---------------------------------------------------------------------------
__global__ __launch_bounds__(128) void gemm_qkv_kernel(
    const unsigned short* __restrict__ X,
    const unsigned short* __restrict__ W,
    const float* __restrict__ bias,
    unsigned short* __restrict__ out, int mode) {

    const int lane  = threadIdx.x & 31;
    const int wave  = threadIdx.x >> 5;
    const int lrow  = lane & 15;
    const int lhalf = lane >> 4;
    const int m0 = blockIdx.y * 128 + wave * 32;
    const int n0 = blockIdx.x * 64;

    v8f acc[2][4];
    #pragma unroll
    for (int tm = 0; tm < 2; ++tm)
        #pragma unroll
        for (int tn = 0; tn < 4; ++tn) acc[tm][tn] = {};

    for (int k0 = 0; k0 < DMODEL; k0 += 32) {
        v16bf a[2], bf[4];
        #pragma unroll
        for (int tm = 0; tm < 2; ++tm)
            a[tm] = load16(X + (size_t)(m0 + tm * 16 + lrow) * DMODEL + k0 + lhalf * 16);
        #pragma unroll
        for (int tn = 0; tn < 4; ++tn)
            bf[tn] = load16(W + (size_t)(n0 + tn * 16 + lrow) * DMODEL + k0 + lhalf * 16);
        #pragma unroll
        for (int tm = 0; tm < 2; ++tm)
            #pragma unroll
            for (int tn = 0; tn < 4; ++tn)
                acc[tm][tn] = wmma_bf16(a[tm], bf[tn], acc[tm][tn]);
    }

    #pragma unroll
    for (int tm = 0; tm < 2; ++tm) {
        #pragma unroll
        for (int tn = 0; tn < 4; ++tn) {
            const int n  = n0 + tn * 16 + lrow;
            const float bv = bias[n];
            const int h  = n >> 6;
            const int dk = n & 63;
            #pragma unroll
            for (int r = 0; r < 8; ++r) {
                const int m = m0 + tm * 16 + r + 8 * lhalf;
                const int b = m >> 11;
                const int s = m & 2047;
                float v = acc[tm][tn][r] + bv;
                if (mode == 2) {
                    out[(size_t)((b * NHEAD + h) * DK + dk) * SLEN + s] = f2bf(v);
                } else {
                    if (mode == 0) v *= 0.125f;  // 1/sqrt(64) folded into Q
                    out[(size_t)((b * NHEAD + h) * SLEN + s) * DK + dk] = f2bf(v);
                }
            }
        }
    }
}

// ---------------------------------------------------------------------------
// TDM descriptor issue: 2-D bf16 tile -> LDS.
//   rows x cols elements, row stride `stride` elems, tile packed dense in LDS.
// Group0: count=1 | lds_addr | global_addr(120:64) | type=2 (127:126)
// Group1: data_size=1(2B) @17:16 | tensor_dim0@79:48 | tensor_dim1@111:80
//         | tile_dim0@127:112 | tile_dim1@143:128 | dim0_stride@207:160
// Groups 2/3 zero (2-D tile). 6-arg clang-23 builtin signature.
// ---------------------------------------------------------------------------
__device__ __forceinline__ void tdm_load_2d(unsigned int lds_addr,
                                            unsigned long long gaddr,
                                            unsigned int cols, unsigned int rows,
                                            unsigned int stride) {
    u32x4 g0 = {};
    g0[0] = 1u;                                     // count = 1, user descriptor
    g0[1] = lds_addr;                               // LDS byte address
    g0[2] = (unsigned int)(gaddr & 0xffffffffull);  // global_addr[31:0]
    g0[3] = (unsigned int)((gaddr >> 32) & 0x1ffffffull) | (2u << 30); // + type=2
    i32x8 g1 = {};
    g1[0] = (int)(1u << 16);                        // data_size = 1 (2 bytes)
    g1[1] = (int)(cols << 16);                      // tensor_dim0[15:0]
    g1[2] = (int)(rows << 16);                      // tensor_dim1[15:0] @ bits 95:80
    g1[3] = (int)(cols << 16);                      // tile_dim0 @ bits 127:112
    g1[4] = (int)rows;                              // tile_dim1 @ bits 143:128
    g1[5] = (int)stride;                            // tensor_dim0_stride[31:0]
    i32x4 z4 = {};
    i32x8 z8 = {};
    __builtin_amdgcn_tensor_load_to_lds(g0, g1, z4, z4, z8, 0);
}

// ---------------------------------------------------------------------------
// Flash attention, one (b,h) per block column.
// grid = (S/128, B*H), block = 256 (8 waves); each wave owns 16 query rows.
// Per 32-key step: TDM stages K tile (32x64) and V^T tile (64x32) into LDS
// (double-buffered, issued by wave 0, s_wait_tensorcnt + barriers); every
// wave then runs 4 QK^T WMMAs, online softmax, P->A transpose via private
// LDS tile, 4 PV WMMAs.
// ---------------------------------------------------------------------------
__global__ __launch_bounds__(256) void flash_attn_kernel(
    const unsigned short* __restrict__ Qp,   // [B*H, S, Dk], pre-scaled
    const unsigned short* __restrict__ Kp,   // [B*H, S, Dk]
    const unsigned short* __restrict__ Vt,   // [B*H, Dk, S]
    unsigned short* __restrict__ ctx) {      // [B, S, D]

    __shared__ unsigned short Klds[2][32 * DK];   // [key][dk]
    __shared__ unsigned short Vlds[2][DK * 32];   // [dk][key]
    __shared__ unsigned short Plds[8][16 * 32];   // per-wave transpose tile

    const int lane  = threadIdx.x & 31;
    const int wave  = threadIdx.x >> 5;
    const int lrow  = lane & 15;
    const int lhalf = lane >> 4;
    const int bh = blockIdx.y;
    const int q0 = blockIdx.x * 128 + wave * 16;

    const unsigned long long kgbase =
        (unsigned long long)(const void*)(Kp + (size_t)bh * SLEN * DK);
    const unsigned long long vgbase =
        (unsigned long long)(const void*)(Vt + (size_t)bh * DK * SLEN);
    const unsigned int kl0 = (unsigned int)(size_t)(void*)&Klds[0][0];
    const unsigned int kl1 = (unsigned int)(size_t)(void*)&Klds[1][0];
    const unsigned int vl0 = (unsigned int)(size_t)(void*)&Vlds[0][0];
    const unsigned int vl1 = (unsigned int)(size_t)(void*)&Vlds[1][0];

    // Q fragments (16x32 A-operands), Dk=64 -> two K-chunks
    const unsigned short* qbase =
        Qp + (size_t)(bh * SLEN + q0 + lrow) * DK + lhalf * 16;
    const v16bf aQ0 = load16(qbase);
    const v16bf aQ1 = load16(qbase + 32);

    float mrun[8], lrun[8];
    v8f acc[4];
    #pragma unroll
    for (int r = 0; r < 8; ++r) { mrun[r] = -3.0e38f; lrun[r] = 0.0f; }
    #pragma unroll
    for (int t = 0; t < 4; ++t) acc[t] = {};

    const int NIT = SLEN / 32;

    // prologue: stage tile 0 into buffer 0
    if (wave == 0) {
        tdm_load_2d(kl0, kgbase,            DK, 32, DK);    // K rows 0..31
        tdm_load_2d(vl0, vgbase,            32, DK, SLEN);  // V^T cols 0..31
    }

    for (int i = 0; i < NIT; ++i) {
        const int buf = i & 1;
        __syncthreads();   // everyone done reading buffer (i+1)&1 from iter i-1
        if (wave == 0) {
            if (i + 1 < NIT) {
                const int nb = (i + 1) & 1;
                const unsigned long long nk = kgbase + (unsigned long long)(i + 1) * 32 * DK * 2;
                const unsigned long long nv = vgbase + (unsigned long long)(i + 1) * 32 * 2;
                tdm_load_2d(nb ? kl1 : kl0, nk, DK, 32, DK);
                tdm_load_2d(nb ? vl1 : vl0, nv, 32, DK, SLEN);
                __builtin_amdgcn_s_wait_tensorcnt((short)2);  // tile i done
            } else {
                __builtin_amdgcn_s_wait_tensorcnt((short)0);
            }
        }
        __syncthreads();   // publish tile i

        // ---- scores for 32 keys (two 16-wide tiles) from LDS
        v8f s0 = {}, s1 = {};
        {
            const unsigned short* kb0 = &Klds[buf][(size_t)lrow * DK + lhalf * 16];
            s0 = wmma_bf16(aQ0, load16(kb0),      s0);
            s0 = wmma_bf16(aQ1, load16(kb0 + 32), s0);
            const unsigned short* kb1 = kb0 + 16 * DK;
            s1 = wmma_bf16(aQ0, load16(kb1),      s1);
            s1 = wmma_bf16(aQ1, load16(kb1 + 32), s1);
        }

        // ---- online softmax (row stats replicated across 16 lanes)
        float alpha[8];
        #pragma unroll
        for (int r = 0; r < 8; ++r) {
            float t = fmaxf(s0[r], s1[r]);
            #pragma unroll
            for (int msk = 1; msk <= 8; msk <<= 1)
                t = fmaxf(t, __shfl_xor(t, msk, 32));
            const float mn = fmaxf(mrun[r], t);
            alpha[r] = __expf(mrun[r] - mn);
            mrun[r] = mn;
            const float p0 = __expf(s0[r] - mn);
            const float p1 = __expf(s1[r] - mn);
            s0[r] = p0; s1[r] = p1;
            float ps = p0 + p1;
            #pragma unroll
            for (int msk = 1; msk <= 8; msk <<= 1)
                ps += __shfl_xor(ps, msk, 32);
            lrun[r] = lrun[r] * alpha[r] + ps;
        }

        // ---- P (C-layout) -> A-layout via wave-private LDS tile.
        // DS ops are in-order within a wave; no block barrier needed.
        #pragma unroll
        for (int r = 0; r < 8; ++r) {
            const int m = r + 8 * lhalf;
            Plds[wave][m * 32 + lrow]      = f2bf(s0[r]);
            Plds[wave][m * 32 + 16 + lrow] = f2bf(s1[r]);
        }
        const v16bf aP = load16(&Plds[wave][lrow * 32 + lhalf * 16]);

        // ---- rescale running accumulators
        #pragma unroll
        for (int t = 0; t < 4; ++t)
            #pragma unroll
            for (int r = 0; r < 8; ++r) acc[t][r] *= alpha[r];

        // ---- PV from LDS V^T tile
        const unsigned short* vb = &Vlds[buf][(size_t)lrow * 32 + lhalf * 16];
        #pragma unroll
        for (int t = 0; t < 4; ++t)
            acc[t] = wmma_bf16(aP, load16(vb + (size_t)t * 16 * 32), acc[t]);
    }

    // ---- normalize and store context
    const int b = bh >> 4;
    const int h = bh & 15;
    #pragma unroll
    for (int t = 0; t < 4; ++t) {
        #pragma unroll
        for (int r = 0; r < 8; ++r) {
            const int m = r + 8 * lhalf;
            const float v = acc[t][r] / lrun[r];
            const int row = b * SLEN + q0 + m;
            const int col = h * DK + t * 16 + lrow;
            ctx[(size_t)row * DMODEL + col] = f2bf(v);
        }
    }
}

// ---------------------------------------------------------------------------
// Output GEMM: out = ctx[4096,1024](bf16) * Wo^T + b_o  (fp32 result)
// ---------------------------------------------------------------------------
__global__ __launch_bounds__(128) void gemm_out_kernel(
    const unsigned short* __restrict__ X,
    const unsigned short* __restrict__ W,
    const float* __restrict__ bias,
    float* __restrict__ out) {

    const int lane  = threadIdx.x & 31;
    const int wave  = threadIdx.x >> 5;
    const int lrow  = lane & 15;
    const int lhalf = lane >> 4;
    const int m0 = blockIdx.y * 128 + wave * 32;
    const int n0 = blockIdx.x * 64;

    v8f acc[2][4];
    #pragma unroll
    for (int tm = 0; tm < 2; ++tm)
        #pragma unroll
        for (int tn = 0; tn < 4; ++tn) acc[tm][tn] = {};

    for (int k0 = 0; k0 < DMODEL; k0 += 32) {
        v16bf a[2], bf[4];
        #pragma unroll
        for (int tm = 0; tm < 2; ++tm)
            a[tm] = load16(X + (size_t)(m0 + tm * 16 + lrow) * DMODEL + k0 + lhalf * 16);
        #pragma unroll
        for (int tn = 0; tn < 4; ++tn)
            bf[tn] = load16(W + (size_t)(n0 + tn * 16 + lrow) * DMODEL + k0 + lhalf * 16);
        #pragma unroll
        for (int tm = 0; tm < 2; ++tm)
            #pragma unroll
            for (int tn = 0; tn < 4; ++tn)
                acc[tm][tn] = wmma_bf16(a[tm], bf[tn], acc[tm][tn]);
    }

    #pragma unroll
    for (int tm = 0; tm < 2; ++tm) {
        #pragma unroll
        for (int tn = 0; tn < 4; ++tn) {
            const int n = n0 + tn * 16 + lrow;
            const float bv = bias[n];
            #pragma unroll
            for (int r = 0; r < 8; ++r) {
                const int m = m0 + tm * 16 + r + 8 * lhalf;
                out[(size_t)m * DMODEL + n] = acc[tm][tn][r] + bv;
            }
        }
    }
}

// ---------------------------------------------------------------------------
// Host-side orchestration
// ---------------------------------------------------------------------------
extern "C" void kernel_launch(void* const* d_in, const int* in_sizes, int n_in,
                              void* d_out, int out_size, void* d_ws, size_t ws_size,
                              hipStream_t stream) {
    const float* q   = (const float*)d_in[0];
    const float* k   = (const float*)d_in[1];
    const float* v   = (const float*)d_in[2];
    const float* w_q = (const float*)d_in[3];
    const float* b_q = (const float*)d_in[4];
    const float* w_k = (const float*)d_in[5];
    const float* b_k = (const float*)d_in[6];
    const float* w_v = (const float*)d_in[7];
    const float* b_v = (const float*)d_in[8];
    const float* w_o = (const float*)d_in[9];
    const float* b_o = (const float*)d_in[10];
    float* out = (float*)d_out;

    // workspace carving (all bf16-as-ushort), ~64 MiB total
    char* ws = (char*)d_ws;
    size_t off = 0;
    auto carve = [&](size_t elems) {
        unsigned short* p = (unsigned short*)(ws + off);
        off += elems * sizeof(unsigned short);
        return p;
    };
    const size_t NX = (size_t)MROWS * DMODEL;   // 4M elems
    const size_t NW = (size_t)DMODEL * DMODEL;  // 1M elems
    unsigned short* xq  = carve(NX);
    unsigned short* xk  = carve(NX);
    unsigned short* xv  = carve(NX);
    unsigned short* wqb = carve(NW);
    unsigned short* wkb = carve(NW);
    unsigned short* wvb = carve(NW);
    unsigned short* wob = carve(NW);
    unsigned short* Qp  = carve(NX);  // [B*H, S, Dk]
    unsigned short* Kp  = carve(NX);  // [B*H, S, Dk]
    unsigned short* Vt  = carve(NX);  // [B*H, Dk, S]
    unsigned short* ctx = carve(NX);  // [B, S, D]

    // 1) fp32 -> bf16 conversions
    cvt_bf16_kernel<<<1024, 256, 0, stream>>>(q,   xq,  (int)NX);
    cvt_bf16_kernel<<<1024, 256, 0, stream>>>(k,   xk,  (int)NX);
    cvt_bf16_kernel<<<1024, 256, 0, stream>>>(v,   xv,  (int)NX);
    cvt_bf16_kernel<<<512,  256, 0, stream>>>(w_q, wqb, (int)NW);
    cvt_bf16_kernel<<<512,  256, 0, stream>>>(w_k, wkb, (int)NW);
    cvt_bf16_kernel<<<512,  256, 0, stream>>>(w_v, wvb, (int)NW);
    cvt_bf16_kernel<<<512,  256, 0, stream>>>(w_o, wob, (int)NW);

    // 2) Q/K/V projections (Q pre-scaled by 1/sqrt(Dk), V stored transposed)
    dim3 ggrid(DMODEL / 64, MROWS / 128);  // (16, 32)
    gemm_qkv_kernel<<<ggrid, 128, 0, stream>>>(xq, wqb, b_q, Qp, 0);
    gemm_qkv_kernel<<<ggrid, 128, 0, stream>>>(xk, wkb, b_k, Kp, 1);
    gemm_qkv_kernel<<<ggrid, 128, 0, stream>>>(xv, wvb, b_v, Vt, 2);

    // 3) flash attention: grid (S/128, B*H), 8 waves/block sharing K/V tiles
    flash_attn_kernel<<<dim3(SLEN / 128, 32), 256, 0, stream>>>(Qp, Kp, Vt, ctx);

    // 4) output projection -> fp32
    gemm_out_kernel<<<ggrid, 128, 0, stream>>>(ctx, wob, b_o, out);
}
